// IGNNConv_87600152969918
// MI455X (gfx1250) — matrix operand order repaired and checked by reading, not compiled, not run on hardware
//
#include <hip/hip_runtime.h>

#define N_NODES 50000
#define N_EDGES 800000
#define IN_F    96
#define H_F     256
#define NHOP    4      // hops 0..3 (inceptive branches)
#define MB      64     // GEMM M-tile per block
#define PH      104    // LDS pitch for h tile (96 + 8 pad -> conflict-free b128)
#define PN      264    // LDS pitch for ns tile (256 + 8 pad)

typedef __attribute__((ext_vector_type(8)))  float  v8f;
typedef __attribute__((ext_vector_type(16))) __bf16 v16bf;

__device__ __forceinline__ unsigned short f2bf(float x) {
  // round-to-nearest-even f32 -> bf16 (bit-level, no __bf16 scalar arithmetic)
  unsigned int u = __float_as_uint(x);
  u += 0x7FFFu + ((u >> 16) & 1u);
  return (unsigned short)(u >> 16);
}

// Async DMA of one 16 KB weight chunk (8192 bf16) global -> LDS.
// 256 threads x 4 x b128; INST_OFFSET applies to BOTH the LDS and global side,
// so one VGPR pair + one LDS-address VGPR covers the chunk.
// Tracked by ASYNCcnt (4 per wave per chunk, completes in order).
__device__ __forceinline__ void async_chunk(const unsigned short* __restrict__ src,
                                            unsigned short* dst, int tid) {
  unsigned lds           = (unsigned)(size_t)dst + (unsigned)(tid * 16);
  unsigned long long gad = (unsigned long long)(size_t)src + (unsigned long long)(tid * 16);
  asm volatile("global_load_async_to_lds_b128 %0, %1, off\n\t"
               "global_load_async_to_lds_b128 %0, %1, off offset:4096\n\t"
               "global_load_async_to_lds_b128 %0, %1, off offset:8192\n\t"
               "global_load_async_to_lds_b128 %0, %1, off offset:12288"
               :: "v"(lds), "v"(gad)
               : "memory");
}

// ---------------------------------------------------------------------------
// Graph preparation: degree count -> scan -> CSR scatter
// ---------------------------------------------------------------------------
__global__ void k_zero2(int* a, int* b, int n) {
  int i = blockIdx.x * blockDim.x + threadIdx.x;
  if (i < n) { a[i] = 0; b[i] = 0; }
}

__global__ void k_count(const int* __restrict__ ei, int* __restrict__ cnt) {
  int e = blockIdx.x * blockDim.x + threadIdx.x;
  if (e < N_EDGES) {
    int r = ei[e], c = ei[N_EDGES + e];
    if (r != c) atomicAdd(&cnt[r], 1);   // pre-existing diagonal entries weigh 0
  }
}

__global__ __launch_bounds__(1024) void k_scan(const int* __restrict__ cnt,
                                               int* __restrict__ rowptr,
                                               float* __restrict__ dinv) {
  __shared__ int buf[1024];
  __shared__ int base_s;
  int tid = threadIdx.x;
  if (tid == 0) base_s = 0;
  __syncthreads();
  for (int start = 0; start < N_NODES; start += 1024) {
    int i = start + tid;
    int v = (i < N_NODES) ? cnt[i] : 0;
    buf[tid] = v;
    __syncthreads();
    for (int off = 1; off < 1024; off <<= 1) {
      int t = (tid >= off) ? buf[tid - off] : 0;
      __syncthreads();
      buf[tid] += t;
      __syncthreads();
    }
    int incl = buf[tid];
    int base = base_s;
    if (i < N_NODES) {
      rowptr[i] = base + incl - v;                 // exclusive prefix
      dinv[i]   = rsqrtf((float)(v + 1));          // deg = offdiag-count + self loop
    }
    __syncthreads();
    if (tid == 1023) base_s = base + incl;
    __syncthreads();
  }
  if (tid == 0) rowptr[N_NODES] = base_s;
}

__global__ void k_scatter(const int* __restrict__ ei, const int* __restrict__ rowptr,
                          int* __restrict__ cnt2, const float* __restrict__ dinv,
                          int* __restrict__ ccol, float* __restrict__ cw) {
  int e = blockIdx.x * blockDim.x + threadIdx.x;
  if (e < N_EDGES) {
    int r = ei[e], c = ei[N_EDGES + e];
    if (r != c) {
      int pos = rowptr[r] + atomicAdd(&cnt2[r], 1);
      ccol[pos] = c;
      cw[pos]   = dinv[r] * dinv[c];
    }
  }
}

// ---------------------------------------------------------------------------
// Row-normalize features (F.normalize dim=1)
// ---------------------------------------------------------------------------
__global__ void k_norm(const float* __restrict__ feat, float* __restrict__ x0) {
  int r = blockIdx.x;
  int f = threadIdx.x;                      // blockDim = 96 (3 waves)
  __shared__ float s[IN_F];
  __shared__ float inv;
  float v = feat[(size_t)r * IN_F + f];
  s[f] = v * v;
  __syncthreads();
  if (f == 0) {
    float t = 0.f;
    #pragma unroll
    for (int j = 0; j < IN_F; ++j) t += s[j];
    inv = 1.0f / fmaxf(sqrtf(t), 1e-12f);
  }
  __syncthreads();
  x0[(size_t)r * IN_F + f] = v * inv;
}

// ---------------------------------------------------------------------------
// CSR SpMM: xout[r] = dinv[r]^2 * xin[r] + sum_e cw[e] * xin[ccol[e]]
// ---------------------------------------------------------------------------
__global__ void k_spmm(const int* __restrict__ rowptr, const int* __restrict__ ccol,
                       const float* __restrict__ cw, const float* __restrict__ dinv,
                       const float* __restrict__ xin, float* __restrict__ xout) {
  int r = blockIdx.x;
  int f = threadIdx.x;                      // blockDim = 96
  float d = dinv[r];
  float acc = d * d * xin[(size_t)r * IN_F + f];
  int e0 = rowptr[r], e1 = rowptr[r + 1];
  for (int e = e0; e < e1; ++e)
    acc += cw[e] * xin[(size_t)ccol[e] * IN_F + f];
  xout[(size_t)r * IN_F + f] = acc;
}

// ---------------------------------------------------------------------------
// Pre-swizzle weights into WMMA B-fragment order (bf16).
// B (32x16) in this code's contiguous k order (A matches): lane = half*16 + n, element e holds K = half*16 + e.
// Fragment f = kt*16 + nt stored contiguously (512 elems).
// ---------------------------------------------------------------------------
__global__ void k_wswiz(const float* __restrict__ Ws, const float* __restrict__ Wf,
                        unsigned short* __restrict__ wsw, unsigned short* __restrict__ wfw) {
  const int T1 = NHOP * IN_F * H_F;    // 98304
  const int T2 = NHOP * H_F * H_F;     // 262144
  for (int idx = blockIdx.x * blockDim.x + threadIdx.x; idx < T1 + T2;
       idx += gridDim.x * blockDim.x) {
    if (idx < T1) {
      int i   = idx / (IN_F * H_F);
      int rem = idx - i * (IN_F * H_F);
      int k = rem / H_F, n = rem & (H_F - 1);
      int kt = k >> 5, kk = k & 31, hf = kk >> 4, e = kk & 15;
      int nt = n >> 4, nn = n & 15;
      int off = ((kt * 16 + nt) * 32 + (hf * 16 + nn)) * 16 + e;
      wsw[i * 24576 + off] = f2bf(Ws[idx]);
    } else {
      int j  = idx - T1;
      int kg = j / H_F, n = j & (H_F - 1);
      int i  = kg >> 8, k = kg & 255;
      int kt = k >> 5, kk = k & 31, hf = kk >> 4, e = kk & 15;
      int nt = n >> 4, nn = n & 15;
      int off = ((kt * 16 + nt) * 32 + (hf * 16 + nn)) * 16 + e;
      wfw[i * 65536 + off] = f2bf(Wf[j]);
    }
  }
}

// ---------------------------------------------------------------------------
// Fused per-node-tile network:
//   out = relu( bf + sum_i relu(h_i @ Ws_i + bs_i) @ Wf[i*256:(i+1)*256] )
// 64 rows/block, 8 waves; wave = (mtile 0..3, nhalf 0..1) -> 16x128 out slice.
// ns never touches HBM; per-wave 16x128 f32 accumulator lives across hops.
// Weight K-chunks stream via async global->LDS DMA, double-buffered so the
// transfer of chunk kt+1 overlaps the 8 WMMAs of chunk kt.
// ---------------------------------------------------------------------------
__global__ __launch_bounds__(256) void k_gemm(
    const float* __restrict__ xall,
    const unsigned short* __restrict__ wsw,
    const unsigned short* __restrict__ wfw,
    const float* __restrict__ bs,
    const float* __restrict__ bfv,
    float* __restrict__ out) {
  __shared__ alignas(16) unsigned short sh_h[MB * PH];      // 13312 B
  __shared__ alignas(16) unsigned short sh_ns[MB * PN];     // 33792 B
  __shared__ alignas(32) unsigned short sh_w0[8192];        // 16384 B chunk buf 0
  __shared__ alignas(32) unsigned short sh_w1[8192];        // 16384 B chunk buf 1

  const int tid   = threadIdx.x;
  const int lane  = tid & 31;
  const int wave  = tid >> 5;
  const int mtile = wave & 3;
  const int nhalf = wave >> 2;
  const int ln    = lane & 15;
  const int half  = lane >> 4;
  const int m0    = blockIdx.x * MB;
  const int arow  = mtile * 16 + ln;   // A-fragment row for this lane

  v8f acc[8] = {};

  for (int hop = 0; hop < NHOP; ++hop) {
    // ---- stage h_i tile: f32 -> bf16 LDS (float4 loads, packed uint2 stores)
    const float* xp = xall + (size_t)hop * (size_t)N_NODES * IN_F;
    #pragma unroll
    for (int q = 0; q < 6; ++q) {
      int idx4 = tid + 256 * q;                 // 0..1535 float4 units
      int r = idx4 / 24, c4 = idx4 - r * 24;
      int gr = m0 + r;
      float4 v = make_float4(0.f, 0.f, 0.f, 0.f);
      if (gr < N_NODES) v = *(const float4*)(xp + (size_t)gr * IN_F + c4 * 4);
      union { unsigned short s[4]; uint2 u; } pk;
      pk.s[0] = f2bf(v.x); pk.s[1] = f2bf(v.y);
      pk.s[2] = f2bf(v.z); pk.s[3] = f2bf(v.w);
      *(uint2*)(sh_h + r * PH + c4 * 4) = pk.u;
    }

    // ---- GEMM1: c1 = h @ Ws[hop]  (K = 96 -> 3 k-steps, double-buffered) ----
    const unsigned short* w1base = wsw + hop * 24576;
    async_chunk(w1base, sh_w0, tid);                       // prologue: chunk 0
    v8f c1[8] = {};
    #pragma unroll
    for (int kt = 0; kt < 3; ++kt) {
      const unsigned short* cur = (kt & 1) ? sh_w1 : sh_w0;
      if (kt + 1 < 3) {
        async_chunk(w1base + (kt + 1) * 8192, (kt & 1) ? sh_w0 : sh_w1, tid);
        asm volatile("s_wait_asynccnt 0x4" ::: "memory");  // chunk kt landed
      } else {
        asm volatile("s_wait_asynccnt 0x0" ::: "memory");
      }
      __syncthreads();          // chunk kt visible to all; staged tile ready (kt==0)
      union { uint4 q[2]; v16bf v; } a;
      const int k0 = kt * 32 + half * 16;
      a.q[0] = *(const uint4*)(sh_h + arow * PH + k0);
      a.q[1] = *(const uint4*)(sh_h + arow * PH + k0 + 8);
      #pragma unroll
      for (int t = 0; t < 8; ++t) {
        int nt = nhalf * 8 + t;
        v16bf b = *(const v16bf*)(cur + (nt * 32 + lane) * 16);
        c1[t] = __builtin_amdgcn_wmma_f32_16x16x32_bf16(false, a.v, false, b,
                                                        (short)0, c1[t], false, false);
      }
      __syncthreads();          // all reads of chunk kt done -> its buffer reusable
    }
    // bias + ReLU -> ns (bf16, LDS only). D layout: M = half*8+j, N = ln.
    #pragma unroll
    for (int t = 0; t < 8; ++t) {
      int col = (nhalf * 8 + t) * 16 + ln;
      float bias = bs[hop * H_F + col];
      #pragma unroll
      for (int j = 0; j < 8; ++j) {
        int row = mtile * 16 + half * 8 + j;
        sh_ns[row * PN + col] = f2bf(fmaxf(c1[t][j] + bias, 0.0f));
      }
    }

    // ---- GEMM2: acc += ns @ Wf[hop]  (K = 256 -> 8 k-steps, double-buffered) ----
    const unsigned short* w2base = wfw + hop * 65536;
    async_chunk(w2base, sh_w0, tid);                       // prologue: chunk 0
    #pragma unroll
    for (int kt = 0; kt < 8; ++kt) {
      const unsigned short* cur = (kt & 1) ? sh_w1 : sh_w0;
      if (kt + 1 < 8) {
        async_chunk(w2base + (kt + 1) * 8192, (kt & 1) ? sh_w0 : sh_w1, tid);
        asm volatile("s_wait_asynccnt 0x4" ::: "memory");
      } else {
        asm volatile("s_wait_asynccnt 0x0" ::: "memory");
      }
      __syncthreads();          // chunk kt visible; kt==0 also fences ns writes
      union { uint4 q[2]; v16bf v; } a;
      const int k0 = kt * 32 + half * 16;
      a.q[0] = *(const uint4*)(sh_ns + arow * PN + k0);
      a.q[1] = *(const uint4*)(sh_ns + arow * PN + k0 + 8);
      #pragma unroll
      for (int t = 0; t < 8; ++t) {
        int nt = nhalf * 8 + t;
        v16bf b = *(const v16bf*)(cur + (nt * 32 + lane) * 16);
        acc[t] = __builtin_amdgcn_wmma_f32_16x16x32_bf16(false, a.v, false, b,
                                                         (short)0, acc[t], false, false);
      }
      __syncthreads();
    }
  }

  // final bias + ReLU -> out (f32)
  #pragma unroll
  for (int t = 0; t < 8; ++t) {
    int col = (nhalf * 8 + t) * 16 + ln;
    float bias = bfv[col];
    #pragma unroll
    for (int j = 0; j < 8; ++j) {
      int grow = m0 + mtile * 16 + half * 8 + j;
      if (grow < N_NODES)
        out[(size_t)grow * H_F + col] = fmaxf(acc[t][j] + bias, 0.0f);
    }
  }
}

// ---------------------------------------------------------------------------
extern "C" void kernel_launch(void* const* d_in, const int* in_sizes, int n_in,
                              void* d_out, int out_size, void* d_ws, size_t ws_size,
                              hipStream_t stream) {
  (void)in_sizes; (void)n_in; (void)out_size; (void)ws_size;
  const int*   ei   = (const int*)  d_in[0];   // edge_index [2, N_EDGES] (int32)
  const float* feat = (const float*)d_in[1];   // [N, 96]
  const float* Ws   = (const float*)d_in[2];   // [4, 96, 256]
  const float* bs   = (const float*)d_in[3];   // [4, 256]
  const float* Wf   = (const float*)d_in[4];   // [1024, 256]
  const float* bfv  = (const float*)d_in[5];   // [256]
  float* out = (float*)d_out;                  // [N, 256]

  char* p = (char*)d_ws;
  auto take = [&](size_t bytes) -> char* {
    char* r = p;
    p += (bytes + 255) & ~(size_t)255;
    return r;
  };
  int*   cnt    = (int*)  take((size_t)N_NODES * sizeof(int));
  int*   cnt2   = (int*)  take((size_t)N_NODES * sizeof(int));
  int*   rowptr = (int*)  take((size_t)(N_NODES + 1) * sizeof(int));
  float* dinv   = (float*)take((size_t)N_NODES * sizeof(float));
  int*   ccol   = (int*)  take((size_t)N_EDGES * sizeof(int));
  float* cw     = (float*)take((size_t)N_EDGES * sizeof(float));
  float* xall   = (float*)take((size_t)NHOP * N_NODES * IN_F * sizeof(float));
  unsigned short* wsw = (unsigned short*)take((size_t)NHOP * 24576 * sizeof(unsigned short));
  unsigned short* wfw = (unsigned short*)take((size_t)NHOP * 65536 * sizeof(unsigned short));

  k_zero2  <<<(N_NODES + 255) / 256, 256, 0, stream>>>(cnt, cnt2, N_NODES);
  k_count  <<<(N_EDGES + 255) / 256, 256, 0, stream>>>(ei, cnt);
  k_scan   <<<1, 1024, 0, stream>>>(cnt, rowptr, dinv);
  k_scatter<<<(N_EDGES + 255) / 256, 256, 0, stream>>>(ei, rowptr, cnt2, dinv, ccol, cw);
  k_norm   <<<N_NODES, IN_F, 0, stream>>>(feat, xall);
  for (int h = 0; h < NHOP - 1; ++h) {
    const float* xin = xall + (size_t)h       * N_NODES * IN_F;
    float*       xo  = xall + (size_t)(h + 1) * N_NODES * IN_F;
    k_spmm<<<N_NODES, IN_F, 0, stream>>>(rowptr, ccol, cw, dinv, xin, xo);
  }
  k_wswiz  <<<512, 256, 0, stream>>>(Ws, Wf, wsw, wfw);
  k_gemm   <<<(N_NODES + MB - 1) / MB, 256, 0, stream>>>(xall, wsw, wfw, bs, bfv, out);
}
